// RPNModule_42683384987716
// MI455X (gfx1250) — matrix operand order repaired and checked by reading, not compiled, not run on hardware
//
#include <hip/hip_runtime.h>
#include <hip/hip_bf16.h>
#include <stdint.h>
#include <stddef.h>

// ---------------------------------------------------------------------------
// RPN head for MI455X (gfx1250): bf16 WMMA implicit-GEMM convs (async-to-LDS
// double-buffered staging) + proposal generation
// ---------------------------------------------------------------------------

typedef __attribute__((ext_vector_type(16))) __bf16        v16bf;
typedef __attribute__((ext_vector_type(8)))  float         v8f;
typedef __attribute__((ext_vector_type(4)))  unsigned int  u32x4;
typedef int v4i_vs __attribute__((vector_size(16)));   // matches builtin sig

#define NIMG   8
#define CIN    512
#define HW     4096          // 64*64
#define K1     4608          // 9*512
#define ANCH   36864         // HW * 9
#define NPAD   65536         // padded pow2 for bitonic sort
#define PRE    2000
#define POST   300

union Frag16 { v16bf v; u32x4 q[2]; };

#ifndef __has_builtin
#define __has_builtin(x) 0
#endif

// ---- CDNA5 async global->LDS copy (ASYNCcnt path) -------------------------
__device__ __forceinline__ void async_load_b128(const void* g, void* l) {
#if __has_builtin(__builtin_amdgcn_global_load_async_to_lds_b128)
  __builtin_amdgcn_global_load_async_to_lds_b128(
      (__attribute__((address_space(1))) v4i_vs*)g,
      (__attribute__((address_space(3))) v4i_vs*)l,
      0, 0);
#else
  unsigned int la = (unsigned int)(unsigned long long)l;
  asm volatile("global_load_async_to_lds_b128 %0, %1, off"
               :: "v"(la), "v"(g) : "memory");
#endif
}

__device__ __forceinline__ void wait_async0() {
#if __has_builtin(__builtin_amdgcn_s_wait_asynccnt)
  __builtin_amdgcn_s_wait_asynccnt(0);
#else
  asm volatile("s_wait_asynccnt 0x0" ::: "memory");
#endif
}

__device__ __forceinline__ unsigned int fsortkey(float f) {
  unsigned int b = __float_as_uint(f);
  return (b & 0x80000000u) ? ~b : (b | 0x80000000u);
}
#define KEY_NEGINF 0x007FFFFFu   // fsortkey(-inf)
#define KEY_POSINF 0xFF800000u   // fsortkey(+inf)

// ---------------------------------------------------------------------------
// Pack kernels: f32 -> bf16, layout changes for implicit GEMM
// ---------------------------------------------------------------------------

// x: NCHW f32 -> NHWC bf16
__global__ void pack_x(const float* __restrict__ x, __bf16* __restrict__ xb) {
  int t = blockIdx.x * 256 + threadIdx.x;            // 8*4096*512 total
  int c = t & 511;
  int p = (t >> 9) & 4095;
  int n = t >> 21;
  xb[t] = (__bf16)x[((size_t)n * 512 + c) * 4096 + p];
}

// conv1 weights OIHW -> w1t[oc][k], k = (ky*3+kx)*512 + c  (k contiguous)
__global__ void pack_w1(const float* __restrict__ w, __bf16* __restrict__ wt) {
  int t  = blockIdx.x * 256 + threadIdx.x;           // 512*4608 total
  int oc = t / K1;
  int k  = t - oc * K1;
  int c  = k & 511;
  int t9 = k >> 9;
  int ky = t9 / 3;
  int kx = t9 - ky * 3;
  wt[t] = (__bf16)w[(size_t)oc * K1 + (size_t)c * 9 + ky * 3 + kx];
}

// loc (36x512) + score (18x512) 1x1 weights -> w2t[col][k], 64 cols padded
__global__ void pack_w2(const float* __restrict__ locw,
                        const float* __restrict__ scow,
                        __bf16* __restrict__ wt) {
  int t   = blockIdx.x * 256 + threadIdx.x;          // 64*512 total
  int col = t >> 9;
  int k   = t & 511;
  float v = 0.0f;
  if (col < 36)       v = locw[(size_t)col * 512 + k];
  else if (col < 54)  v = scow[(size_t)(col - 36) * 512 + k];
  wt[t] = (__bf16)v;
}

// ---------------------------------------------------------------------------
// 3x3 conv 512->512 + bias + ReLU as implicit GEMM (bf16 WMMA, f32 accum)
// M = 32768 (N*H*W), N = 512 (oc), K = 4608.  Tile 128x128x32, 8 waves/WG.
// Double-buffered LDS, async global->LDS staging, one barrier per k-step.
// ---------------------------------------------------------------------------

// stage one 128x32 A tile + 128x32 B tile (per-thread: 32B A + 32B B)
__device__ __forceinline__ void stage_tile(
    const __bf16* __restrict__ xb, const __bf16* __restrict__ w1t,
    int img, int n0, int sy, int sx, int srow, int sseg, int kt,
    __bf16 (*As)[32], __bf16 (*Bs)[32]) {
  const int t9 = kt >> 4;                  // which of 9 taps
  const int c0 = (kt & 15) << 5;           // channel base within tap
  const int ky = t9 / 3;
  const int kx = t9 - ky * 3;
  const int yy = sy + ky - 1;
  const int xx = sx + kx - 1;

  __bf16* adst = &As[srow][sseg * 16];
  if ((unsigned)yy < 64u && (unsigned)xx < 64u) {
    const __bf16* src =
        xb + (((size_t)img * 64 + yy) * 64 + xx) * 512 + c0 + sseg * 16;
    async_load_b128(src,     adst);
    async_load_b128(src + 8, adst + 8);
  } else {
    const u32x4 z = {0, 0, 0, 0};
    *(u32x4*)(adst)     = z;               // DScnt path, covered by barrier
    *(u32x4*)(adst + 8) = z;
  }
  const __bf16* bsrc = w1t + (size_t)(n0 + srow) * K1 + kt * 32 + sseg * 16;
  __bf16* bdst = &Bs[srow][sseg * 16];
  async_load_b128(bsrc,     bdst);
  async_load_b128(bsrc + 8, bdst + 8);
  __builtin_prefetch((const void*)(bsrc + 32), 0, 1);   // next k-chunk of B
}

__global__ __launch_bounds__(256)
void conv3x3_wmma(const __bf16* __restrict__ xb,
                  const __bf16* __restrict__ w1t,
                  const float*  __restrict__ bias,
                  __bf16* __restrict__ hb) {
  __shared__ __bf16 As[2][128][32];   // [buf][spatial row][k]
  __shared__ __bf16 Bs[2][128][32];   // [buf][oc col][k]

  const int n0   = blockIdx.x * 128;       // out-channel tile (4 tiles)
  const int m0   = blockIdx.y * 128;       // spatial tile (256 tiles)
  const int img  = m0 >> 12;
  const int y0   = (m0 & 4095) >> 6;       // tile = two full image rows

  const int tid  = threadIdx.x;
  const int lane = tid & 31;
  const int wv   = tid >> 5;
  const int wm   = wv & 3;                 // 4 groups of 32 rows
  const int wn   = wv >> 2;                // 2 groups of 64 cols
  const int half = lane >> 4;
  const int l15  = lane & 15;

  const v8f zero8 = {0.f,0.f,0.f,0.f,0.f,0.f,0.f,0.f};
  v8f acc[2][4];
  #pragma unroll
  for (int i = 0; i < 2; ++i)
    #pragma unroll
    for (int j = 0; j < 4; ++j) acc[i][j] = zero8;

  const int srow = tid >> 1;               // 0..127
  const int sseg = tid & 1;                // 16 bf16 elements each
  const int sy   = y0 + (srow >> 6);
  const int sx   = srow & 63;

  // prologue: stage k-chunk 0 into buffer 0
  stage_tile(xb, w1t, img, n0, sy, sx, srow, sseg, 0, As[0], Bs[0]);
  wait_async0();
  __syncthreads();

  for (int kt = 0; kt < K1 / 32; ++kt) {
    const int cur = kt & 1;
    const int nxt = cur ^ 1;
    if (kt + 1 < K1 / 32)
      stage_tile(xb, w1t, img, n0, sy, sx, srow, sseg, kt + 1,
                 As[nxt], Bs[nxt]);

    // fragments per ISA 16-bit A(16x32)/B(32x16) lane layouts
    Frag16 afr[2], bfr[4];
    #pragma unroll
    for (int mf = 0; mf < 2; ++mf) {
      const int row = wm * 32 + mf * 16 + l15;
      const int klo = half * 8;   // lanes 0-15: K0-7/16-23, 16-31: K8-15/24-31
      afr[mf].q[0] = *(const u32x4*)&As[cur][row][klo];
      afr[mf].q[1] = *(const u32x4*)&As[cur][row][klo + 16];
    }
    #pragma unroll
    for (int nf = 0; nf < 4; ++nf) {
      const int col = wn * 64 + nf * 16 + l15;
      const int ko  = half * 8;   // lanes 0-15: K0-7/16-23, 16-31: K8-15/24-31
      bfr[nf].q[0] = *(const u32x4*)&Bs[cur][col][ko];
      bfr[nf].q[1] = *(const u32x4*)&Bs[cur][col][ko + 16];
    }
    #pragma unroll
    for (int mf = 0; mf < 2; ++mf)
      #pragma unroll
      for (int nf = 0; nf < 4; ++nf)
        acc[mf][nf] = __builtin_amdgcn_wmma_f32_16x16x32_bf16(
            false, afr[mf].v, false, bfr[nf].v,
            (short)0, acc[mf][nf], false, false);

    wait_async0();                 // next buffer fully landed in LDS
    __syncthreads();               // (also waits DScnt for zero-fill stores)
  }

  // epilogue: bias + ReLU -> NHWC bf16
  #pragma unroll
  for (int mf = 0; mf < 2; ++mf)
    #pragma unroll
    for (int nf = 0; nf < 4; ++nf) {
      const int colb = n0 + wn * 64 + nf * 16 + l15;
      const float bv = bias[colb];
      #pragma unroll
      for (int r = 0; r < 8; ++r) {
        const int mlocal = wm * 32 + mf * 16 + r + half * 8;
        float v = acc[mf][nf][r] + bv;
        v = v > 0.f ? v : 0.f;
        hb[(size_t)(m0 + mlocal) * 512 + colb] = (__bf16)v;
      }
    }
}

// ---------------------------------------------------------------------------
// Fused 1x1 convs (loc 36 + score 18 -> 64 padded cols) as bf16 WMMA GEMM
// M = 32768, K = 512, Ncols = 64.  Tile 128x64x32, 8 waves (32x32 each).
// NHWC GEMM output == reference's transpose(0,2,3,1) layout.
// ---------------------------------------------------------------------------
__global__ __launch_bounds__(256)
void conv1x1_wmma(const __bf16* __restrict__ hb,
                  const __bf16* __restrict__ w2t,
                  const float*  __restrict__ loc_b,
                  const float*  __restrict__ sco_b,
                  float* __restrict__ out_locs,
                  float* __restrict__ out_scores) {
  __shared__ __bf16 As[128][32];
  __shared__ __bf16 Bs[64][32];

  const int m0   = blockIdx.x * 128;
  const int tid  = threadIdx.x;
  const int lane = tid & 31;
  const int wv   = tid >> 5;
  const int wm   = wv & 3;                 // 32-row groups
  const int wn   = wv >> 2;                // 32-col groups
  const int half = lane >> 4;
  const int l15  = lane & 15;

  const v8f zero8 = {0.f,0.f,0.f,0.f,0.f,0.f,0.f,0.f};
  v8f acc[2][2];
  #pragma unroll
  for (int i = 0; i < 2; ++i)
    #pragma unroll
    for (int j = 0; j < 2; ++j) acc[i][j] = zero8;

  const int srow = tid >> 1;
  const int sseg = tid & 1;

  for (int kt = 0; kt < 16; ++kt) {
    const __bf16* asrc = hb + (size_t)(m0 + srow) * 512 + kt * 32 + sseg * 16;
    u32x4 a0 = *(const u32x4*)(asrc);
    u32x4 a1 = *(const u32x4*)(asrc + 8);
    u32x4 b0 = {0,0,0,0}, b1 = {0,0,0,0};
    if (tid < 128) {
      const __bf16* bsrc = w2t + (size_t)srow * 512 + kt * 32 + sseg * 16;
      b0 = *(const u32x4*)(bsrc);
      b1 = *(const u32x4*)(bsrc + 8);
    }
    __syncthreads();
    *(u32x4*)&As[srow][sseg * 16]     = a0;
    *(u32x4*)&As[srow][sseg * 16 + 8] = a1;
    if (tid < 128) {
      *(u32x4*)&Bs[srow][sseg * 16]     = b0;
      *(u32x4*)&Bs[srow][sseg * 16 + 8] = b1;
    }
    __syncthreads();

    Frag16 afr[2], bfr[2];
    #pragma unroll
    for (int mf = 0; mf < 2; ++mf) {
      const int row = wm * 32 + mf * 16 + l15;
      const int klo = half * 8;
      afr[mf].q[0] = *(const u32x4*)&As[row][klo];
      afr[mf].q[1] = *(const u32x4*)&As[row][klo + 16];
    }
    #pragma unroll
    for (int nf = 0; nf < 2; ++nf) {
      const int col = wn * 32 + nf * 16 + l15;
      const int ko  = half * 8;
      bfr[nf].q[0] = *(const u32x4*)&Bs[col][ko];
      bfr[nf].q[1] = *(const u32x4*)&Bs[col][ko + 16];
    }
    #pragma unroll
    for (int mf = 0; mf < 2; ++mf)
      #pragma unroll
      for (int nf = 0; nf < 2; ++nf)
        acc[mf][nf] = __builtin_amdgcn_wmma_f32_16x16x32_bf16(
            false, afr[mf].v, false, bfr[nf].v,
            (short)0, acc[mf][nf], false, false);
  }

  #pragma unroll
  for (int mf = 0; mf < 2; ++mf)
    #pragma unroll
    for (int nf = 0; nf < 2; ++nf) {
      const int col = wn * 32 + nf * 16 + l15;      // 0..63
      #pragma unroll
      for (int r = 0; r < 8; ++r) {
        const int mg = m0 + wm * 32 + mf * 16 + r + half * 8;  // global row
        const float v = acc[mf][nf][r];
        if (col < 36) {
          out_locs[(size_t)mg * 36 + col] = v + loc_b[col];
        } else if (col < 54) {
          const int c2 = col - 36;
          out_scores[(size_t)mg * 18 + c2] = v + sco_b[c2];
        }
      }
    }
}

// ---------------------------------------------------------------------------
// Decode anchors + locs -> clipped boxes, min-size filter, build sort keys
// ---------------------------------------------------------------------------
__global__ void decode_boxes(const float* __restrict__ locs,
                             const float* __restrict__ scores,
                             const int* __restrict__ ph,
                             const int* __restrict__ pw,
                             float* __restrict__ boxes,
                             unsigned long long* __restrict__ keys) {
  const int t    = blockIdx.x * 256 + threadIdx.x;  // 8 * 65536
  const int img  = t >> 16;
  const int slot = t & 65535;
  unsigned long long key;
  if (slot < ANCH) {
    const int pos = slot / 9;
    const int a   = slot - pos * 9;
    const int y   = pos >> 6;
    const int x   = pos & 63;
    const int ri  = a / 3;
    const int si  = a - ri * 3;
    const float ratios[3] = {0.5f, 1.0f, 2.0f};
    const float scales[3] = {8.0f, 16.0f, 32.0f};
    const float r  = ratios[ri];
    const float s  = scales[si];
    const float ha = 16.0f * s * sqrtf(r);
    const float wa = 16.0f * s * sqrtf(1.0f / r);
    const float acy = (float)y * 16.0f + 8.0f;
    const float acx = (float)x * 16.0f + 8.0f;

    const float* lp = locs + (size_t)(img * 4096 + pos) * 36 + a * 4;
    const float dy = lp[0], dx = lp[1], dh = lp[2], dw = lp[3];
    const float cy = dy * ha + acy;
    const float cx = dx * wa + acx;
    const float hh = expf(dh) * ha;
    const float wwv = expf(dw) * wa;

    const float fh = (float)ph[0];
    const float fw = (float)pw[0];
    const float ymin = fminf(fmaxf(cy - 0.5f * hh, 0.f), fh);
    const float xmin = fminf(fmaxf(cx - 0.5f * wwv, 0.f), fw);
    const float ymax = fminf(fmaxf(cy + 0.5f * hh, 0.f), fh);
    const float xmax = fminf(fmaxf(cx + 0.5f * wwv, 0.f), fw);

    const bool valid = ((ymax - ymin) >= 16.0f) && ((xmax - xmin) >= 16.0f);
    const float fg = scores[(size_t)img * 73728 + (size_t)pos * 18 + a * 2 + 1];
    const float sc = valid ? fg : -INFINITY;

    float* bp = boxes + ((size_t)img * ANCH + slot) * 4;
    bp[0] = ymin; bp[1] = xmin; bp[2] = ymax; bp[3] = xmax;
    key = ((unsigned long long)fsortkey(sc) << 32) | (unsigned)slot;
  } else {
    key = ((unsigned long long)KEY_NEGINF << 32) | (unsigned)slot;
  }
  keys[t] = key;
}

// ---------------------------------------------------------------------------
// Per-image descending bitonic sort of 65536 keys (1 workgroup per image)
// ---------------------------------------------------------------------------
__global__ __launch_bounds__(1024)
void bitonic_sort(unsigned long long* __restrict__ keys) {
  unsigned long long* a = keys + (size_t)blockIdx.x * NPAD;
  for (int k = 2; k <= NPAD; k <<= 1) {
    for (int j = k >> 1; j > 0; j >>= 1) {
      for (int i = threadIdx.x; i < NPAD; i += 1024) {
        const int ixj = i ^ j;
        if (ixj > i) {
          const unsigned long long va = a[i];
          const unsigned long long vb = a[ixj];
          const bool up = ((i & k) == 0);       // descending overall
          if (up ? (va < vb) : (va > vb)) { a[i] = vb; a[ixj] = va; }
        }
      }
      __threadfence_block();
      __syncthreads();
    }
  }
}

// ---------------------------------------------------------------------------
// NMS over top-2000 per image + compaction to 300 ROIs (1 WG per image)
// ---------------------------------------------------------------------------
__global__ __launch_bounds__(1024)
void nms_kernel(const unsigned long long* __restrict__ keys,
                const float* __restrict__ boxes,
                float* __restrict__ rois,
                int* __restrict__ roi_idx) {
  __shared__ float bx[PRE][4];
  __shared__ unsigned char keep[PRE];
  __shared__ int sel[POST];
  __shared__ int s_cnt;

  const int img = blockIdx.x;
  const unsigned long long* k = keys + (size_t)img * NPAD;

  for (int j = threadIdx.x; j < PRE; j += 1024) {
    const unsigned long long key = k[j];
    const unsigned int s    = (unsigned int)(key >> 32);
    const unsigned int slot = (unsigned int)(key & 0xFFFFFFFFull);
    const bool fin = (s > KEY_NEGINF) && (s < KEY_POSINF);
    keep[j] = fin ? 1 : 0;
    if (fin && slot < ANCH) {
      const float* bp = boxes + ((size_t)img * ANCH + slot) * 4;
      bx[j][0] = bp[0]; bx[j][1] = bp[1]; bx[j][2] = bp[2]; bx[j][3] = bp[3];
    } else {
      bx[j][0] = 0.f; bx[j][1] = 0.f; bx[j][2] = 0.f; bx[j][3] = 0.f;
    }
  }
  __syncthreads();

  for (int i = 0; i < PRE - 1; ++i) {
    if (keep[i]) {
      const float ay = bx[i][0], ax = bx[i][1];
      const float ay2 = bx[i][2], ax2 = bx[i][3];
      const float aarea = (ay2 - ay) * (ax2 - ax);
      for (int j = i + 1 + threadIdx.x; j < PRE; j += 1024) {
        if (keep[j]) {
          const float ty  = fmaxf(ay,  bx[j][0]);
          const float tx  = fmaxf(ax,  bx[j][1]);
          const float by  = fminf(ay2, bx[j][2]);
          const float bxr = fminf(ax2, bx[j][3]);
          const float inter = fmaxf(by - ty, 0.f) * fmaxf(bxr - tx, 0.f);
          const float barea = (bx[j][2] - bx[j][0]) * (bx[j][3] - bx[j][1]);
          const float iou = inter / (aarea + barea - inter + 1e-9f);
          if (iou > 0.7f) keep[j] = 0;
        }
      }
    }
    __syncthreads();
  }

  if (threadIdx.x == 0) {
    int c = 0;
    for (int i = 0; i < PRE && c < POST; ++i)
      if (keep[i]) sel[c++] = i;
    s_cnt = c;
  }
  __syncthreads();
  const int cnt = s_cnt;
  for (int r = threadIdx.x; r < POST; r += 1024) {
    float v0 = 0.f, v1 = 0.f, v2 = 0.f, v3 = 0.f;
    if (r < cnt) {
      const int i = sel[r];
      v0 = bx[i][0]; v1 = bx[i][1]; v2 = bx[i][2]; v3 = bx[i][3];
    }
    float* rp = rois + ((size_t)img * POST + r) * 4;
    rp[0] = v0; rp[1] = v1; rp[2] = v2; rp[3] = v3;
    roi_idx[img * POST + r] = img;
  }
}

// ---------------------------------------------------------------------------
// Launcher
// ---------------------------------------------------------------------------
extern "C" void kernel_launch(void* const* d_in, const int* in_sizes, int n_in,
                              void* d_out, int out_size, void* d_ws, size_t ws_size,
                              hipStream_t stream) {
  (void)in_sizes; (void)n_in; (void)out_size; (void)ws_size;

  const float* x       = (const float*)d_in[0];
  const float* w1      = (const float*)d_in[1];
  const float* b1      = (const float*)d_in[2];
  const float* score_w = (const float*)d_in[3];
  const float* score_b = (const float*)d_in[4];
  const float* loc_w   = (const float*)d_in[5];
  const float* loc_b   = (const float*)d_in[6];
  const int*   img_h   = (const int*)d_in[7];
  const int*   img_w   = (const int*)d_in[8];

  // workspace layout (bytes)
  char* ws = (char*)d_ws;
  __bf16* xb   = (__bf16*)(ws);                         // 33,554,432
  __bf16* w1t  = (__bf16*)(ws + 33554432);              //  4,718,592
  __bf16* hb   = (__bf16*)(ws + 38273024);              // 33,554,432
  __bf16* w2t  = (__bf16*)(ws + 71827456);              //     65,536
  float*  boxes = (float*)(ws + 71892992);              //  4,718,592
  unsigned long long* keys = (unsigned long long*)(ws + 76611584); // 4,194,304

  // output layout (f32 elements; roi_indices region holds int32 bits)
  float* out        = (float*)d_out;
  float* out_locs   = out;                  // 8*36864*4 = 1,179,648
  float* out_scores = out + 1179648;        // 8*36864*2 =   589,824
  float* out_rois   = out + 1769472;        // 8*300*4   =     9,600
  int*   out_ridx   = (int*)(out + 1779072);// 8*300     =     2,400

  pack_x <<<65536, 256, 0, stream>>>(x, xb);
  pack_w1<<<9216,  256, 0, stream>>>(w1, w1t);
  pack_w2<<<128,   256, 0, stream>>>(loc_w, score_w, w2t);

  conv3x3_wmma<<<dim3(4, 256), 256, 0, stream>>>(xb, w1t, b1, hb);
  conv1x1_wmma<<<256, 256, 0, stream>>>(hb, w2t, loc_b, score_b,
                                        out_locs, out_scores);

  decode_boxes<<<2048, 256, 0, stream>>>(out_locs, out_scores, img_h, img_w,
                                         boxes, keys);
  bitonic_sort<<<NIMG, 1024, 0, stream>>>(keys);
  nms_kernel  <<<NIMG, 1024, 0, stream>>>(keys, boxes, out_rois, out_ridx);
}
